// Pointnet2_6244882448518
// MI455X (gfx1250) — matrix operand order repaired and checked by reading, not compiled, not run on hardware
//
#include <hip/hip_runtime.h>

typedef __attribute__((ext_vector_type(16))) _Float16 v16h;
typedef __attribute__((ext_vector_type(8)))  _Float16 v8h;
typedef __attribute__((ext_vector_type(8)))  float    v8f;

#define EPS_ 1e-5f
#define E1_ 524288   // 8*1024*64 edges, SA1
#define E2_ 131072   // 8*256*64 edges, SA2
#define M1_ 8192     // 8*1024 centers
#define M2_ 2048     // 8*256 centers

// ---------------------------------------------------------------------------
// Blocked WMMA GEMM with async-LDS weight staging.
//   Y[M,N] = relu(X[M,Kp] * Wt[N,Kp]^T + bias), f16 out.
// Block = 256 threads = 8 waves. Block (bx,by) covers rows [bx*128, bx*128+128)
// and columns [by*64, by*64+64). The 64xKp f16 weight strip is copied to LDS
// once per block via global_load_async_to_lds_b128 (ASYNCcnt), then each wave
// computes one 16-row, K-accumulated x 64-column output strip: 4 accumulators reusing
// a single A fragment per K-step (4 WMMAs per A load clause).
// Requires: M % 128 == 0, N % 64 == 0, Kp % 32 == 0.
// ---------------------------------------------------------------------------
__global__ void gemm_f16_lds_kernel(const _Float16* __restrict__ X,
                                    const _Float16* __restrict__ Wt,
                                    const float* __restrict__ bias,
                                    _Float16* __restrict__ Y,
                                    int M, int N, int Kp)
{
    extern __shared__ _Float16 smem[];            // [64][Kp] weight strip

    const int wave = threadIdx.x >> 5;
    const int lane = threadIdx.x & 31;
    const int l15  = lane & 15;
    const int hi   = (lane >> 4) & 1;
    const int strip = blockIdx.y;                 // N-strip of 64 columns

    // ---- async copy of weight strip [64, Kp] (contiguous) into LDS ----
    {
        const unsigned ldsBase = (unsigned)(size_t)(void*)smem;   // LDS byte addr
        const char* gsrc = (const char*)(Wt + (size_t)strip * 64 * Kp);
        const int chunks = 8 * Kp;                // (64*Kp*2)/16 16B chunks
        for (int t = threadIdx.x; t < chunks; t += 256) {
            unsigned lds = ldsBase + (unsigned)(t * 16);
            const char* g = gsrc + (size_t)t * 16;
            asm volatile("global_load_async_to_lds_b128 %0, %1, off"
                         :: "v"(lds), "v"(g) : "memory");
        }
        asm volatile("s_wait_asynccnt 0x0" ::: "memory");
    }
    __syncthreads();

    // ---- compute: one M-tile per wave, 4 N-tiles, K stepped by 32 ----
    const int rowA = blockIdx.x * 128 + wave * 16 + l15;
    const _Float16* xrow = X + (size_t)rowA * Kp;

    v8f acc0 = {}, acc1 = {}, acc2 = {}, acc3 = {};
    for (int k0 = 0; k0 < Kp; k0 += 32) {
        // A 16x32 f16 fragment (shared across the 4 WMMAs below)
        int kA = k0 + hi * 16;
        v16h a = *(const v16h*)(xrow + kA);
        __builtin_prefetch(xrow + kA + 64, 0, 0);

        // B fragments from LDS: lane holds strip-column j*16+l15,
        // 16 contiguous K halves at k0 (+16 for upper half-wave).
        const int kB = k0 + hi * 16;
        v16h b0 = *(const v16h*)(smem + (size_t)(0 * 16 + l15) * Kp + kB);
        v16h b1 = *(const v16h*)(smem + (size_t)(1 * 16 + l15) * Kp + kB);
        v16h b2 = *(const v16h*)(smem + (size_t)(2 * 16 + l15) * Kp + kB);
        v16h b3 = *(const v16h*)(smem + (size_t)(3 * 16 + l15) * Kp + kB);

        acc0 = __builtin_amdgcn_wmma_f32_16x16x32_f16(false, a, false, b0, (short)0, acc0, false, false);
        acc1 = __builtin_amdgcn_wmma_f32_16x16x32_f16(false, a, false, b1, (short)0, acc1, false, false);
        acc2 = __builtin_amdgcn_wmma_f32_16x16x32_f16(false, a, false, b2, (short)0, acc2, false, false);
        acc3 = __builtin_amdgcn_wmma_f32_16x16x32_f16(false, a, false, b3, (short)0, acc3, false, false);
    }

    // ---- epilogue: bias + relu, f16 stores ----
    const int mBase = blockIdx.x * 128 + wave * 16 + hi * 8;
#pragma unroll
    for (int j = 0; j < 4; ++j) {
        const v8f& acc = (j == 0) ? acc0 : (j == 1) ? acc1 : (j == 2) ? acc2 : acc3;
        int   n  = strip * 64 + j * 16 + l15;
        float bb = bias[n];
#pragma unroll
        for (int r = 0; r < 8; ++r) {
            float v = fmaxf(acc[r] + bb, 0.0f);
            Y[(size_t)(mBase + r) * N + n] = (_Float16)v;
        }
    }
}

// ---------------------------------------------------------------------------
// Simple WMMA GEMM (classifier head, M = 16): f16 out, optional relu.
// ---------------------------------------------------------------------------
__global__ void gemm_f16_kernel(const _Float16* __restrict__ X,
                                const _Float16* __restrict__ Wt,
                                const float* __restrict__ bias,
                                _Float16* __restrict__ Y,
                                int M, int N, int Kp, int relu)
{
    int wave = blockIdx.x * (blockDim.x >> 5) + (threadIdx.x >> 5);
    int lane = threadIdx.x & 31;
    int tilesN = N >> 4;
    int tiles  = (M >> 4) * tilesN;
    if (wave >= tiles) return;               // uniform per wave
    int tm = wave / tilesN, tn = wave % tilesN;
    int l15 = lane & 15;
    int hi  = (lane >> 4) & 1;

    const _Float16* xrow = X  + (size_t)(tm * 16 + l15) * Kp;
    const _Float16* wrow = Wt + (size_t)(tn * 16 + l15) * Kp;

    v8f acc = {};
    for (int k0 = 0; k0 < Kp; k0 += 32) {
        int kA = k0 + hi * 16;
        v16h a = *(const v16h*)(xrow + kA);
        v16h b = *(const v16h*)(wrow + k0 + hi * 16);
        acc = __builtin_amdgcn_wmma_f32_16x16x32_f16(false, a, false, b, (short)0, acc, false, false);
    }

    int   n  = tn * 16 + l15;
    float bb = bias ? bias[n] : 0.0f;
#pragma unroll
    for (int r = 0; r < 8; ++r) {
        int   m = tm * 16 + r + hi * 8;
        float v = acc[r] + bb;
        if (relu) v = fmaxf(v, 0.0f);
        Y[(size_t)m * N + n] = (_Float16)v;
    }
}

// Final classifier layer: f32 output, no relu, valid-row guard.
__global__ void gemm_f32out_kernel(const _Float16* __restrict__ X,
                                   const _Float16* __restrict__ Wt,
                                   const float* __restrict__ bias,
                                   float* __restrict__ Y,
                                   int M, int N, int Kp, int Mvalid)
{
    int wave = blockIdx.x * (blockDim.x >> 5) + (threadIdx.x >> 5);
    int lane = threadIdx.x & 31;
    int tilesN = N >> 4;
    int tiles  = (M >> 4) * tilesN;
    if (wave >= tiles) return;
    int tm = wave / tilesN, tn = wave % tilesN;
    int l15 = lane & 15;
    int hi  = (lane >> 4) & 1;

    const _Float16* xrow = X  + (size_t)(tm * 16 + l15) * Kp;
    const _Float16* wrow = Wt + (size_t)(tn * 16 + l15) * Kp;

    v8f acc = {};
    for (int k0 = 0; k0 < Kp; k0 += 32) {
        int kA = k0 + hi * 16;
        v16h a = *(const v16h*)(xrow + kA);
        v16h b = *(const v16h*)(wrow + k0 + hi * 16);
        acc = __builtin_amdgcn_wmma_f32_16x16x32_f16(false, a, false, b, (short)0, acc, false, false);
    }

    int   n  = tn * 16 + l15;
    float bb = bias ? bias[n] : 0.0f;
#pragma unroll
    for (int r = 0; r < 8; ++r) {
        int m = tm * 16 + r + hi * 8;
        if (m < Mvalid) Y[(size_t)m * N + n] = acc[r] + bb;
    }
}

// ---------------------------------------------------------------------------
// Weight prep: f32 W[I,O] -> f16 Wt[O,Kp] (transposed, zero-padded K).
// ---------------------------------------------------------------------------
__global__ void prep_w_kernel(const float* __restrict__ w, _Float16* __restrict__ wt,
                              int I, int O, int Kp)
{
    int i = blockIdx.x * blockDim.x + threadIdx.x;
    if (i >= O * Kp) return;
    int o = i / Kp, k = i % Kp;
    wt[i] = (k < I) ? (_Float16)w[(size_t)k * O + o] : (_Float16)0.0f;
}

__global__ void zero_f32_kernel(float* p, int n)
{
    int i = blockIdx.x * blockDim.x + threadIdx.x;
    if (i < n) p[i] = 0.0f;
}

// ---------------------------------------------------------------------------
// BatchNorm batch stats: per-channel sum / sumsq over all rows (atomics).
// ---------------------------------------------------------------------------
__global__ void stats_kernel(const _Float16* __restrict__ Y,
                             float* __restrict__ sum, float* __restrict__ sq,
                             int M, int C, int rowsPerBlock)
{
    int rbase = blockIdx.x * rowsPerBlock;
    int rend  = rbase + rowsPerBlock;
    if (rend > M) rend = M;
    for (int c = threadIdx.x; c < C; c += blockDim.x) {
        float s = 0.0f, q = 0.0f;
        for (int r = rbase; r < rend; ++r) {
            float v = (float)Y[(size_t)r * C + c];
            s += v; q += v * v;
        }
        atomicAdd(&sum[c], s);
        atomicAdd(&sq[c],  q);
    }
}

__global__ void bn_kernel(_Float16* __restrict__ Y,
                          const float* __restrict__ sum, const float* __restrict__ sq,
                          const float* __restrict__ g,   const float* __restrict__ be,
                          int M, int C)
{
    size_t i = (size_t)blockIdx.x * blockDim.x + threadIdx.x;
    if (i >= (size_t)M * C) return;
    int   c   = (int)(i % C);
    float inv = 1.0f / (float)M;
    float mu  = sum[c] * inv;
    float var = sq[c] * inv - mu * mu;
    float v   = (float)Y[i];
    Y[i] = (_Float16)(g[c] * (v - mu) * rsqrtf(var + EPS_) + be[c]);
}

// ---------------------------------------------------------------------------
// SA1 edge-feature gather: H[e,0:9] = [x_j(6), pos_j - c_i(3)], pad to 32.
// ---------------------------------------------------------------------------
__global__ void gather1_kernel(const float* __restrict__ x, const float* __restrict__ pos,
                               const int* __restrict__ idx1, const int* __restrict__ neigh1,
                               _Float16* __restrict__ H, float* __restrict__ c1)
{
    int e = blockIdx.x * blockDim.x + threadIdx.x;
    if (e >= E1_) return;
    int m = e >> 6, k = e & 63;
    int ctr = idx1[m];
    float cx = pos[ctr * 3 + 0], cy = pos[ctr * 3 + 1], cz = pos[ctr * 3 + 2];
    if (k == 0) { c1[m * 3 + 0] = cx; c1[m * 3 + 1] = cy; c1[m * 3 + 2] = cz; }
    int j = neigh1[e];
    _Float16* h = H + (size_t)e * 32;
#pragma unroll
    for (int f = 0; f < 6; ++f) h[f] = (_Float16)x[(size_t)j * 6 + f];
    h[6] = (_Float16)(pos[j * 3 + 0] - cx);
    h[7] = (_Float16)(pos[j * 3 + 1] - cy);
    h[8] = (_Float16)(pos[j * 3 + 2] - cz);
#pragma unroll
    for (int f = 9; f < 32; ++f) h[f] = (_Float16)0.0f;
}

// SA2 edge-feature gather: H[e,0:131] = [x1_j(128), c1_j - c2_i(3)], pad to 160.
__global__ void gather2_kernel(const _Float16* __restrict__ x1, const float* __restrict__ c1,
                               const int* __restrict__ idx2, const int* __restrict__ neigh2,
                               _Float16* __restrict__ H, float* __restrict__ c2)
{
    int e = blockIdx.x * blockDim.x + threadIdx.x;
    if (e >= E2_) return;
    int m = e >> 6, k = e & 63;
    int ctr = idx2[m];
    float cx = c1[ctr * 3 + 0], cy = c1[ctr * 3 + 1], cz = c1[ctr * 3 + 2];
    if (k == 0) { c2[m * 3 + 0] = cx; c2[m * 3 + 1] = cy; c2[m * 3 + 2] = cz; }
    int j = neigh2[e];
    _Float16*       h  = H  + (size_t)e * 160;
    const _Float16* xr = x1 + (size_t)j * 128;
#pragma unroll
    for (int f = 0; f < 16; ++f)
        *(v8h*)(h + f * 8) = *(const v8h*)(xr + f * 8);
    h[128] = (_Float16)(c1[j * 3 + 0] - cx);
    h[129] = (_Float16)(c1[j * 3 + 1] - cy);
    h[130] = (_Float16)(c1[j * 3 + 2] - cz);
#pragma unroll
    for (int f = 131; f < 160; ++f) h[f] = (_Float16)0.0f;
}

// MLP3 input: [x2(256), c2(3)] per row, pad to 288.
__global__ void concat3_kernel(const _Float16* __restrict__ x2, const float* __restrict__ c2,
                               _Float16* __restrict__ H)
{
    int r = blockIdx.x * blockDim.x + threadIdx.x;
    if (r >= M2_) return;
    _Float16*       h  = H  + (size_t)r * 288;
    const _Float16* xr = x2 + (size_t)r * 256;
#pragma unroll
    for (int f = 0; f < 32; ++f)
        *(v8h*)(h + f * 8) = *(const v8h*)(xr + f * 8);
    h[256] = (_Float16)c2[r * 3 + 0];
    h[257] = (_Float16)c2[r * 3 + 1];
    h[258] = (_Float16)c2[r * 3 + 2];
#pragma unroll
    for (int f = 259; f < 288; ++f) h[f] = (_Float16)0.0f;
}

// Max over K=64 neighbors: out[m,c] = max_k Y[(m*64+k)*C + c].
__global__ void pool_k_kernel(const _Float16* __restrict__ Y, _Float16* __restrict__ out,
                              int M, int C)
{
    int i = blockIdx.x * blockDim.x + threadIdx.x;
    if (i >= M * C) return;
    int m = i / C, c = i % C;
    const _Float16* p = Y + ((size_t)m * 64) * C + c;
    float best = -3.0e38f;
    for (int k = 0; k < 64; ++k) best = fmaxf(best, (float)p[(size_t)k * C]);
    out[i] = (_Float16)best;
}

// Global max pool over 256 rows per cloud: g[16,1024], rows 8..15 zeroed.
__global__ void pool_global_kernel(const _Float16* __restrict__ H, _Float16* __restrict__ g)
{
    int i = blockIdx.x * blockDim.x + threadIdx.x;
    if (i >= 16 * 1024) return;
    int r = i >> 10, c = i & 1023;
    if (r >= 8) { g[i] = (_Float16)0.0f; return; }
    const _Float16* p = H + ((size_t)r * 256) * 1024 + c;
    float best = -3.0e38f;
    for (int j = 0; j < 256; ++j) best = fmaxf(best, (float)p[(size_t)j * 1024]);
    g[i] = (_Float16)best;
}

// ---------------------------------------------------------------------------
extern "C" void kernel_launch(void* const* d_in, const int* in_sizes, int n_in,
                              void* d_out, int out_size, void* d_ws, size_t ws_size,
                              hipStream_t stream)
{
    const float* x      = (const float*)d_in[0];
    const float* pos    = (const float*)d_in[1];
    const int*   idx1   = (const int*)d_in[2];
    const int*   neigh1 = (const int*)d_in[3];
    const int*   idx2   = (const int*)d_in[4];
    const int*   neigh2 = (const int*)d_in[5];

    // ---- carve workspace (256B aligned) ----
    char*  ws  = (char*)d_ws;
    size_t off = 0;
    auto take = [&](size_t bytes) -> char* {
        size_t o = (off + 255) & ~(size_t)255;
        off = o + bytes;
        return ws + o;
    };
    _Float16* Abuf = (_Float16*)take((size_t)E1_ * 64 * 2);   //  64 MB ping
    _Float16* Bbuf = (_Float16*)take((size_t)E1_ * 128 * 2);  // 128 MB pong
    float*    c1   = (float*)take((size_t)M1_ * 3 * 4);
    float*    c2   = (float*)take((size_t)M2_ * 3 * 4);
    _Float16* x1   = (_Float16*)take((size_t)M1_ * 128 * 2);
    _Float16* x2   = (_Float16*)take((size_t)M2_ * 256 * 2);
    _Float16* gbuf = (_Float16*)take(16 * 1024 * 2);
    _Float16* gl1  = (_Float16*)take(16 * 512 * 2);
    _Float16* gl2  = (_Float16*)take(16 * 256 * 2);
    float*    ssum = (float*)take(1024 * 4);
    float*    ssq  = (float*)take(1024 * 4);

    // ---- weight descriptors (d_in order: w, b, gamma, beta per MLP layer) ----
    struct LD { int wi, I, O; };
    const LD MLPL[9] = {{6,9,64},{10,64,64},{14,64,128},
                        {18,131,128},{22,128,128},{26,128,256},
                        {30,259,256},{34,256,512},{38,512,1024}};
    const LD LINL[3] = {{42,1024,512},{44,512,256},{46,256,32}};

    _Float16* wtm[9]; int kpm[9];
    for (int i = 0; i < 9; ++i) {
        kpm[i] = ((MLPL[i].I + 31) / 32) * 32;
        wtm[i] = (_Float16*)take((size_t)MLPL[i].O * kpm[i] * 2);
        int n = MLPL[i].O * kpm[i];
        prep_w_kernel<<<(n + 255) / 256, 256, 0, stream>>>(
            (const float*)d_in[MLPL[i].wi], wtm[i], MLPL[i].I, MLPL[i].O, kpm[i]);
    }
    _Float16* wtl[3];
    for (int i = 0; i < 3; ++i) {
        wtl[i] = (_Float16*)take((size_t)LINL[i].O * LINL[i].I * 2);
        int n = LINL[i].O * LINL[i].I;
        prep_w_kernel<<<(n + 255) / 256, 256, 0, stream>>>(
            (const float*)d_in[LINL[i].wi], wtl[i], LINL[i].I, LINL[i].O, LINL[i].I);
    }

    auto gemmBig = [&](const _Float16* X, const _Float16* Wt, const float* bias,
                       _Float16* Y, int M, int N, int Kp) {
        dim3 grid(M / 128, N / 64);
        size_t shmem = (size_t)64 * Kp * 2;        // weight strip, max 64 KB
        gemm_f16_lds_kernel<<<grid, 256, shmem, stream>>>(X, Wt, bias, Y, M, N, Kp);
    };
    auto runLayer = [&](const _Float16* X, int li, _Float16* Y, int M) {
        const LD& L = MLPL[li];
        gemmBig(X, wtm[li], (const float*)d_in[L.wi + 1], Y, M, L.O, kpm[li]);
        zero_f32_kernel<<<(L.O + 255) / 256, 256, 0, stream>>>(ssum, L.O);
        zero_f32_kernel<<<(L.O + 255) / 256, 256, 0, stream>>>(ssq,  L.O);
        stats_kernel<<<(M + 255) / 256, 256, 0, stream>>>(Y, ssum, ssq, M, L.O, 256);
        size_t tot = (size_t)M * L.O;
        bn_kernel<<<(unsigned)((tot + 255) / 256), 256, 0, stream>>>(
            Y, ssum, ssq, (const float*)d_in[L.wi + 2], (const float*)d_in[L.wi + 3], M, L.O);
    };

    // ---- SA1: 524288 edges, 9 -> 64 -> 64 -> 128, max over K ----
    gather1_kernel<<<(E1_ + 255) / 256, 256, 0, stream>>>(x, pos, idx1, neigh1, Abuf, c1);
    runLayer(Abuf, 0, Bbuf, E1_);                  // [E1,32] -> [E1,64]
    runLayer(Bbuf, 1, Abuf, E1_);                  // [E1,64] -> [E1,64]
    runLayer(Abuf, 2, Bbuf, E1_);                  // [E1,64] -> [E1,128]
    pool_k_kernel<<<(M1_ * 128 + 255) / 256, 256, 0, stream>>>(Bbuf, x1, M1_, 128);

    // ---- SA2: 131072 edges, 131 -> 128 -> 128 -> 256, max over K ----
    gather2_kernel<<<(E2_ + 255) / 256, 256, 0, stream>>>(x1, c1, idx2, neigh2, Abuf, c2);
    runLayer(Abuf, 3, Bbuf, E2_);                  // [E2,160] -> [E2,128]
    runLayer(Bbuf, 4, Abuf, E2_);                  // -> [E2,128]
    runLayer(Abuf, 5, Bbuf, E2_);                  // -> [E2,256]
    pool_k_kernel<<<(M2_ * 256 + 255) / 256, 256, 0, stream>>>(Bbuf, x2, M2_, 256);

    // ---- MLP3: 2048 rows, 259 -> 256 -> 512 -> 1024, global max pool ----
    concat3_kernel<<<(M2_ + 255) / 256, 256, 0, stream>>>(x2, c2, Abuf);
    runLayer(Abuf, 6, Bbuf, M2_);                  // [2048,288] -> [2048,256]
    runLayer(Bbuf, 7, Abuf, M2_);                  // -> [2048,512]
    runLayer(Abuf, 8, Bbuf, M2_);                  // -> [2048,1024]
    pool_global_kernel<<<(16 * 1024 + 255) / 256, 256, 0, stream>>>(Bbuf, gbuf);

    // ---- classifier head (M padded 8 -> 16 for WMMA tiles) ----
    {
        int tiles1 = (16 / 16) * (512 / 16);
        gemm_f16_kernel<<<(tiles1 + 7) / 8, 256, 0, stream>>>(
            gbuf, wtl[0], (const float*)d_in[43], gl1, 16, 512, 1024, 1);
        int tiles2 = (16 / 16) * (256 / 16);
        gemm_f16_kernel<<<(tiles2 + 7) / 8, 256, 0, stream>>>(
            gl1, wtl[1], (const float*)d_in[45], gl2, 16, 256, 512, 1);
        int tiles3 = (16 / 16) * (32 / 16);
        gemm_f32out_kernel<<<(tiles3 + 7) / 8, 256, 0, stream>>>(
            gl2, wtl[2], (const float*)d_in[47], (float*)d_out, 16, 32, 256, 8);
    }
}
